// MultiheadAttention_62491774157404
// MI455X (gfx1250) — matrix operand order repaired and checked by reading, not compiled, not run on hardware
//
#include <hip/hip_runtime.h>

// ---- problem constants (reference: BS=4, N_CTX=2048, WIDTH=1024, HEADS=16) ----
#define BS_  4
#define SEQ  2048
#define WID  1024
#define NH   16
#define HD   64
#define N3W  (3*WID)

typedef __attribute__((ext_vector_type(16))) __bf16 v16bf;
typedef __attribute__((ext_vector_type(8)))  __bf16 v8bf;
typedef __attribute__((ext_vector_type(8)))  float  v8f;

static __device__ __forceinline__ v8bf ldv8(const __bf16* p) { return *(const v8bf*)p; }
static __device__ __forceinline__ void stv8(__bf16* p, v8bf v) { *(v8bf*)p = v; }

static __device__ __forceinline__ v16bf combine16(v8bf lo, v8bf hi) {
    v16bf r;
#pragma unroll
    for (int i = 0; i < 8; ++i) { r[i] = lo[i]; r[i + 8] = hi[i]; }
    return r;
}

static __device__ __forceinline__ v8f wmma_bf16(v16bf a, v16bf b, v8f c) {
    // D(16x16 f32) = A(16x32 bf16) * B(32x16 bf16) + C
    return __builtin_amdgcn_wmma_f32_16x16x32_bf16(false, a, false, b, (short)0, c,
                                                   false, false);
}

static __device__ __forceinline__ v8bf cvt8(float4 a, float4 b) {
    v8bf c;
    c[0] = (__bf16)a.x; c[1] = (__bf16)a.y; c[2] = (__bf16)a.z; c[3] = (__bf16)a.w;
    c[4] = (__bf16)b.x; c[5] = (__bf16)b.y; c[6] = (__bf16)b.z; c[7] = (__bf16)b.w;
    return c;
}

// =====================================================================
// Kernel 1: qkv = x @ Wqkv + bqkv  -> scatter to Q/K/V [B,H,S,D] (bf16)
// block = 128 threads (4 waves), tile = 128(M) x 64(N), K-step = 32
// each wave: 32 rows -> 2 A-frags reused over 4 B-frags = 8 WMMA/K-step
// =====================================================================
__global__ __launch_bounds__(128) void qkv_kernel(const float* __restrict__ x,
                                                  const float* __restrict__ Wqkv,
                                                  const float* __restrict__ bqkv,
                                                  __bf16* __restrict__ qb,
                                                  __bf16* __restrict__ kb,
                                                  __bf16* __restrict__ vb) {
    __shared__ __bf16 At[128][40];  // [m][k], stride 40 keeps 16B chunk alignment
    __shared__ __bf16 Bt[64][40];   // [n][k] (W tile transposed)

    const int tid  = threadIdx.x;
    const int wv   = tid >> 5, lane = tid & 31;
    const int ln   = lane & 15, hi = (lane >> 4) & 1;
    const int n0   = blockIdx.x * 64, m0 = blockIdx.y * 128;
    const int bkk  = tid & 31, bng = tid >> 5;           // B tile loader mapping
    const int kbo  = hi ? 16 : 0;                        // A-frag chunk base
    const int kb2  = hi ? 16 : 0;                        // B-frag chunk base

    v8f acc[2][4];
#pragma unroll
    for (int h2 = 0; h2 < 2; ++h2)
#pragma unroll
        for (int t = 0; t < 4; ++t) acc[h2][t] = (v8f){0, 0, 0, 0, 0, 0, 0, 0};

    for (int k0 = 0; k0 < WID; k0 += 32) {
        __syncthreads();
        {   // A tile: 128x32 fp32 -> bf16 LDS; thread tid owns one row (128B)
            const float4* a4 = (const float4*)(x + (size_t)(m0 + tid) * WID + k0);
            stv8(&At[tid][0],  cvt8(a4[0], a4[1]));
            stv8(&At[tid][8],  cvt8(a4[2], a4[3]));
            stv8(&At[tid][16], cvt8(a4[4], a4[5]));
            stv8(&At[tid][24], cvt8(a4[6], a4[7]));
        }
        {   // W tile: 32x64 fp32, stored transposed [n][k] in bf16
            const float4* w4 = (const float4*)(Wqkv + (size_t)(k0 + bkk) * N3W + n0 + bng * 16);
            float4 f0 = w4[0], f1 = w4[1], f2 = w4[2], f3 = w4[3];
            float tmp[16] = {f0.x, f0.y, f0.z, f0.w, f1.x, f1.y, f1.z, f1.w,
                             f2.x, f2.y, f2.z, f2.w, f3.x, f3.y, f3.z, f3.w};
#pragma unroll
            for (int j = 0; j < 16; ++j) Bt[bng * 16 + j][bkk] = (__bf16)tmp[j];
        }
        if (k0 + 32 < WID) {  // prefetch next K-step tiles into cache
            __builtin_prefetch((const void*)(x + (size_t)(m0 + tid) * WID + k0 + 32), 0, 0);
            __builtin_prefetch((const void*)(Wqkv + (size_t)(k0 + 32 + bkk) * N3W + n0 + bng * 16), 0, 0);
        }
        __syncthreads();

        const v16bf af0 = combine16(ldv8(&At[wv * 32 + ln][kbo]),
                                    ldv8(&At[wv * 32 + ln][kbo + 8]));
        const v16bf af1 = combine16(ldv8(&At[wv * 32 + 16 + ln][kbo]),
                                    ldv8(&At[wv * 32 + 16 + ln][kbo + 8]));
#pragma unroll
        for (int t = 0; t < 4; ++t) {
            const v16bf bf = combine16(ldv8(&Bt[t * 16 + ln][kb2]),
                                       ldv8(&Bt[t * 16 + ln][kb2 + 8]));
            acc[0][t] = wmma_bf16(af0, bf, acc[0][t]);
            acc[1][t] = wmma_bf16(af1, bf, acc[1][t]);
        }
    }

    // epilogue: bias + scatter into Q/K/V [B,H,S,D].
    // h/sel/dbase derived from wave-uniform nt (16-col tile never crosses a
    // 64/192 boundary) -> SALU, no per-lane integer division.
#pragma unroll
    for (int t = 0; t < 4; ++t) {
        const int nt    = n0 + t * 16;                   // wave-uniform
        const int h     = nt / 192, rem = nt % 192;
        const int sel   = rem / 64, dbase = rem % 64;
        const int d     = dbase + ln;
        __bf16* dst = (sel == 0) ? qb : (sel == 1) ? kb : vb;
        const float bias = bqkv[nt + ln];
#pragma unroll
        for (int h2 = 0; h2 < 2; ++h2) {
#pragma unroll
            for (int r = 0; r < 8; ++r) {
                const int m = m0 + wv * 32 + h2 * 16 + r + hi * 8;
                const int b = m >> 11, s = m & (SEQ - 1);
                dst[(((size_t)b * NH + h) * SEQ + s) * (size_t)HD + d] =
                    (__bf16)(acc[h2][t][r] + bias);
            }
        }
    }
}

// =====================================================================
// Kernel 2: flash attention per (b,h). block = 4 waves x 16 queries,
// loop over 32-key tiles; online softmax; out -> [B,S,W] bf16
// =====================================================================
__global__ __launch_bounds__(128) void attn_kernel(const __bf16* __restrict__ qbuf,
                                                   const __bf16* __restrict__ kbuf,
                                                   const __bf16* __restrict__ vbuf,
                                                   __bf16* __restrict__ obuf) {
    __shared__ __bf16 Kt[32][72];        // [key][d]   (B-frags for Q*K^T)
    __shared__ __bf16 Vt[64][40];        // [d][key]   (B-frags for P*V)
    __shared__ __bf16 Pt[4][16][40];     // per-wave P tile [qrow][key]

    const int tid  = threadIdx.x;
    const int wv   = tid >> 5, lane = tid & 31;
    const int ln   = lane & 15, hi = (lane >> 4) & 1;
    const int bh   = blockIdx.y;                          // b*NH + h
    const int q0   = blockIdx.x * 64 + wv * 16;
    const size_t base = (size_t)bh * SEQ * HD;
    const int kbo  = hi ? 16 : 0, kb2 = hi ? 16 : 0;
    const float scale = 0.125f;                           // 1/sqrt(64)

    // Q fragments (A layout), d split 0..31 / 32..63
    const __bf16* qp = qbuf + base + (size_t)(q0 + ln) * HD;
    const v16bf aq0 = combine16(ldv8(qp + kbo),      ldv8(qp + kbo + 8));
    const v16bf aq1 = combine16(ldv8(qp + 32 + kbo), ldv8(qp + 32 + kbo + 8));

    v8f acc[4];
#pragma unroll
    for (int t = 0; t < 4; ++t) acc[t] = (v8f){0, 0, 0, 0, 0, 0, 0, 0};
    float mrow[8], lrow[8];
#pragma unroll
    for (int r = 0; r < 8; ++r) { mrow[r] = -3.0e38f; lrow[r] = 0.0f; }

    const int lr = tid >> 2, seg = tid & 3;               // cooperative tile loaders

    for (int kt = 0; kt < SEQ / 32; ++kt) {
        __syncthreads();
        {   // K tile 32x64 -> LDS row-major
            const __bf16* kp = kbuf + base + (size_t)(kt * 32 + lr) * HD + seg * 16;
            stv8(&Kt[lr][seg * 16],     ldv8(kp));
            stv8(&Kt[lr][seg * 16 + 8], ldv8(kp + 8));
        }
        {   // V tile 32x64 -> LDS transposed [d][key]
            const __bf16* vp = vbuf + base + (size_t)(kt * 32 + lr) * HD + seg * 16;
            const v8bf a = ldv8(vp), b = ldv8(vp + 8);
#pragma unroll
            for (int j = 0; j < 8; ++j) {
                Vt[seg * 16 + j][lr]     = a[j];
                Vt[seg * 16 + 8 + j][lr] = b[j];
            }
        }
        if (kt + 1 < SEQ / 32) {  // prefetch next key tile
            __builtin_prefetch((const void*)(kbuf + base + (size_t)((kt + 1) * 32 + lr) * HD + seg * 16), 0, 0);
            __builtin_prefetch((const void*)(vbuf + base + (size_t)((kt + 1) * 32 + lr) * HD + seg * 16), 0, 0);
        }
        __syncthreads();

        // scores: two 16x16 tiles (keys 0-15, 16-31), each accumulated over d=64
        v8f s0 = (v8f){0, 0, 0, 0, 0, 0, 0, 0};
        v8f s1 = s0;
        {
            const v16bf b00 = combine16(ldv8(&Kt[ln][kb2]),      ldv8(&Kt[ln][kb2 + 8]));
            const v16bf b01 = combine16(ldv8(&Kt[ln][32 + kb2]), ldv8(&Kt[ln][32 + kb2 + 8]));
            s0 = wmma_bf16(aq0, b00, s0);
            s0 = wmma_bf16(aq1, b01, s0);
            const v16bf b10 = combine16(ldv8(&Kt[16 + ln][kb2]),      ldv8(&Kt[16 + ln][kb2 + 8]));
            const v16bf b11 = combine16(ldv8(&Kt[16 + ln][32 + kb2]), ldv8(&Kt[16 + ln][32 + kb2 + 8]));
            s1 = wmma_bf16(aq0, b10, s1);
            s1 = wmma_bf16(aq1, b11, s1);
        }

        // online softmax per C-layout row (r + hi*8), reduce across 16-lane col group
#pragma unroll
        for (int r = 0; r < 8; ++r) {
            const float a = s0[r] * scale;
            const float b = s1[r] * scale;
            float mx = fmaxf(a, b);
#pragma unroll
            for (int off = 1; off < 16; off <<= 1) mx = fmaxf(mx, __shfl_xor(mx, off, 32));
            const float mnew  = fmaxf(mrow[r], mx);
            const float alpha = __expf(mrow[r] - mnew);
            const float e0 = __expf(a - mnew);
            const float e1 = __expf(b - mnew);
            float rs = e0 + e1;
#pragma unroll
            for (int off = 1; off < 16; off <<= 1) rs += __shfl_xor(rs, off, 32);
            lrow[r] = lrow[r] * alpha + rs;
            mrow[r] = mnew;
            acc[0][r] *= alpha; acc[1][r] *= alpha; acc[2][r] *= alpha; acc[3][r] *= alpha;
            Pt[wv][r + hi * 8][ln]      = (__bf16)e0;
            Pt[wv][r + hi * 8][16 + ln] = (__bf16)e1;
        }
        // same-wave LDS RAW across lanes: DS is in-order per wave, fence the compiler+DScnt
        asm volatile("s_wait_dscnt 0" ::: "memory");

        // P (A layout, K=32 keys) times V (B layout)
        const v16bf pf = combine16(ldv8(&Pt[wv][ln][kbo]), ldv8(&Pt[wv][ln][kbo + 8]));
#pragma unroll
        for (int t = 0; t < 4; ++t) {
            const v16bf bv = combine16(ldv8(&Vt[t * 16 + ln][kb2]),
                                       ldv8(&Vt[t * 16 + ln][kb2 + 8]));
            acc[t] = wmma_bf16(pf, bv, acc[t]);
        }
    }

    // normalize and write [B,S,W] bf16
    const int b_ = bh >> 4, h_ = bh & 15;
#pragma unroll
    for (int r = 0; r < 8; ++r) {
        const int row = q0 + r + hi * 8;
        const float inv = 1.0f / lrow[r];
        const size_t o = ((size_t)b_ * SEQ + row) * WID + h_ * HD + ln;
        obuf[o]      = (__bf16)(acc[0][r] * inv);
        obuf[o + 16] = (__bf16)(acc[1][r] * inv);
        obuf[o + 32] = (__bf16)(acc[2][r] * inv);
        obuf[o + 48] = (__bf16)(acc[3][r] * inv);
    }
}

// =====================================================================
// Kernel 3: out = attn_out(bf16) @ Wproj + bproj  (fp32 out)
// tile = 128(M) x 64(N), same wave structure as kernel 1
// =====================================================================
__global__ __launch_bounds__(128) void proj_kernel(const __bf16* __restrict__ ab,
                                                   const float* __restrict__ Wp,
                                                   const float* __restrict__ bp,
                                                   float* __restrict__ out) {
    __shared__ __bf16 At[128][40];
    __shared__ __bf16 Bt[64][40];

    const int tid  = threadIdx.x;
    const int wv   = tid >> 5, lane = tid & 31;
    const int ln   = lane & 15, hi = (lane >> 4) & 1;
    const int n0   = blockIdx.x * 64, m0 = blockIdx.y * 128;
    const int bkk  = tid & 31, bng = tid >> 5;
    const int kbo  = hi ? 16 : 0, kb2 = hi ? 16 : 0;

    v8f acc[2][4];
#pragma unroll
    for (int h2 = 0; h2 < 2; ++h2)
#pragma unroll
        for (int t = 0; t < 4; ++t) acc[h2][t] = (v8f){0, 0, 0, 0, 0, 0, 0, 0};

    for (int k0 = 0; k0 < WID; k0 += 32) {
        __syncthreads();
        {   // A tile: 128x32 bf16; thread tid owns one row (64B)
            const __bf16* ap = ab + (size_t)(m0 + tid) * WID + k0;
            stv8(&At[tid][0],  ldv8(ap));
            stv8(&At[tid][8],  ldv8(ap + 8));
            stv8(&At[tid][16], ldv8(ap + 16));
            stv8(&At[tid][24], ldv8(ap + 24));
        }
        {   const float4* w4 = (const float4*)(Wp + (size_t)(k0 + bkk) * WID + n0 + bng * 16);
            float4 f0 = w4[0], f1 = w4[1], f2 = w4[2], f3 = w4[3];
            float tmp[16] = {f0.x, f0.y, f0.z, f0.w, f1.x, f1.y, f1.z, f1.w,
                             f2.x, f2.y, f2.z, f2.w, f3.x, f3.y, f3.z, f3.w};
#pragma unroll
            for (int j = 0; j < 16; ++j) Bt[bng * 16 + j][bkk] = (__bf16)tmp[j];
        }
        if (k0 + 32 < WID) {
            __builtin_prefetch((const void*)(ab + (size_t)(m0 + tid) * WID + k0 + 32), 0, 0);
            __builtin_prefetch((const void*)(Wp + (size_t)(k0 + 32 + bkk) * WID + n0 + bng * 16), 0, 0);
        }
        __syncthreads();

        const v16bf af0 = combine16(ldv8(&At[wv * 32 + ln][kbo]),
                                    ldv8(&At[wv * 32 + ln][kbo + 8]));
        const v16bf af1 = combine16(ldv8(&At[wv * 32 + 16 + ln][kbo]),
                                    ldv8(&At[wv * 32 + 16 + ln][kbo + 8]));
#pragma unroll
        for (int t = 0; t < 4; ++t) {
            const v16bf bf = combine16(ldv8(&Bt[t * 16 + ln][kb2]),
                                       ldv8(&Bt[t * 16 + ln][kb2 + 8]));
            acc[0][t] = wmma_bf16(af0, bf, acc[0][t]);
            acc[1][t] = wmma_bf16(af1, bf, acc[1][t]);
        }
    }

#pragma unroll
    for (int t = 0; t < 4; ++t) {
        const int nn = n0 + t * 16 + ln;
        const float bias = bp[nn];
#pragma unroll
        for (int h2 = 0; h2 < 2; ++h2) {
#pragma unroll
            for (int r = 0; r < 8; ++r) {
                const int m = m0 + wv * 32 + h2 * 16 + r + hi * 8;
                out[(size_t)m * WID + nn] = acc[h2][t][r] + bias;
            }
        }
    }
}

// =====================================================================
extern "C" void kernel_launch(void* const* d_in, const int* in_sizes, int n_in,
                              void* d_out, int out_size, void* d_ws, size_t ws_size,
                              hipStream_t stream) {
    const float* x     = (const float*)d_in[0];
    const float* Wqkv  = (const float*)d_in[1];
    const float* bqkv  = (const float*)d_in[2];
    const float* Wproj = (const float*)d_in[3];
    const float* bproj = (const float*)d_in[4];
    float* out = (float*)d_out;

    // workspace: Q,K,V [B,H,S,D] bf16 (16 MiB each) + attn_out [B,S,W] bf16
    char* ws = (char*)d_ws;
    const size_t BUF = (size_t)BS_ * NH * SEQ * HD * sizeof(__bf16);  // 16 MiB
    __bf16* qb = (__bf16*)(ws);
    __bf16* kb = (__bf16*)(ws + BUF);
    __bf16* vb = (__bf16*)(ws + 2 * BUF);
    __bf16* ob = (__bf16*)(ws + 3 * BUF);

    qkv_kernel<<<dim3(N3W / 64, (BS_ * SEQ) / 128), 128, 0, stream>>>(x, Wqkv, bqkv, qb, kb, vb);
    attn_kernel<<<dim3(SEQ / 64, BS_ * NH), 128, 0, stream>>>(qb, kb, vb, ob);
    proj_kernel<<<dim3(WID / 64, (BS_ * SEQ) / 128), 128, 0, stream>>>(ob, Wproj, bproj, out);
}
